// CrossModalAttention_36215164240753
// MI455X (gfx1250) — matrix operand order repaired and checked by reading, not compiled, not run on hardware
//
#include <hip/hip_runtime.h>
#include <hip/hip_bf16.h>

// ---------------------------------------------------------------------------
// CDNA5 / gfx1250 cross-modal attention.
//   bf16 WMMA (v_wmma_f32_16x16x32_bf16) for all 4 projections + QK^T + PV.
//   GEMM: async global->LDS (ASYNCcnt) double-buffered weight panel,
//         register-pipelined A fragments, peeled epilogue (branchless hot loop).
//   Attention: LDS-resident 16x1024 score panel per workgroup (64KB of the
//         320KB WGP pool) so exact softmax probabilities exist for attn_avg.
// ---------------------------------------------------------------------------

typedef __attribute__((ext_vector_type(16))) __bf16 v16bf;
typedef __attribute__((ext_vector_type(8)))  float  v8f;
typedef __attribute__((ext_vector_type(4)))  int    i4;

union Frag32 { v16bf v; i4 q[2]; };          // one 32-byte WMMA operand fragment
union Pack8  { unsigned short s[8]; i4 q; }; // 8 bf16 -> one 128-bit store

// ---- gfx1250 async global->LDS copy path (ASYNCcnt) -----------------------
#if defined(__has_builtin)
#if __has_builtin(__builtin_amdgcn_global_load_async_to_lds_b128) && \
    __has_builtin(__builtin_amdgcn_s_wait_asynccnt)
#define CMA_HAS_ASYNC 1
#endif
#endif

#ifdef CMA_HAS_ASYNC
// builtin signature (from hipcc diagnostic): (v4i __device__* src,
//                                            v4i __shared__* dst, imm, cpol)
#define CMA_ASYNC_COPY16(dst, src)                                        \
  __builtin_amdgcn_global_load_async_to_lds_b128(                         \
      (__attribute__((address_space(1))) i4*)(void*)(src),                \
      (__attribute__((address_space(3))) i4*)(void*)(dst), 0, 0)
#define CMA_ASYNC_WAIT(n) __builtin_amdgcn_s_wait_asynccnt(n)
#else
#define CMA_ASYNC_COPY16(dst, src) (*(i4*)(dst) = *(const i4*)(src))
#define CMA_ASYNC_WAIT(n) ((void)0)
#endif

__device__ __forceinline__ unsigned short f32_to_bf16(float f) {
  unsigned int u = __float_as_uint(f);
  u += 0x7FFFu + ((u >> 16) & 1u);           // round-to-nearest-even
  return (unsigned short)(u >> 16);
}

__device__ __forceinline__ v8f wmma_bf16(v16bf a, v16bf b, v8f c) {
  return __builtin_amdgcn_wmma_f32_16x16x32_bf16(false, a, false, b,
                                                 (short)0, c, false, false);
}

// ---------------------------------------------------------------------------
// f32 -> bf16 bulk convert (tokens + weights)
// ---------------------------------------------------------------------------
__global__ void cma_cvt_bf16(const float* __restrict__ in,
                             unsigned short* __restrict__ out, int n) {
  int i = blockIdx.x * blockDim.x + threadIdx.x;
  int stride = gridDim.x * blockDim.x;
  for (; i < n; i += stride) out[i] = f32_to_bf16(in[i]);
}

// ---------------------------------------------------------------------------
// C[M,N] = A[M,K] * W[N,K]^T + bias    (both operands K-major bf16)
//   mode 0: store bf16 row-major [M,N]            (Q, K projections)
//   mode 1: store bf16 transposed Vt[b][h][d][p]  (V projection, N=768=H*D)
//   mode 2: store f32 row-major [M,N]             (final O projection -> d_out)
// Workgroup = 8 waves, each wave owns a 16x64 tile -> WG tile 128x64.
// Weight panel [64n x 64k] is async-copied to LDS (double buffered, shared by
// all 8 waves); A fragments are per-wave-private and register-pipelined.
// ---------------------------------------------------------------------------
__global__ __launch_bounds__(256)
void cma_gemm_tn(const unsigned short* __restrict__ A,
                 const unsigned short* __restrict__ W,
                 const float* __restrict__ bias,
                 void* __restrict__ out,
                 int M, int N, int K, int mode) {
  __shared__ unsigned short sW[2][64 * 64];   // 2 x 8KB weight panels

  const int lane = threadIdx.x & 31;
  const int wave = threadIdx.x >> 5;
  const int mt   = blockIdx.x * 128 + wave * 16;
  const int nb   = blockIdx.y * 64;
  const int l16  = lane & 15;
  const int aoff = (lane < 16) ? 0 : 8;    // A fragment k-chunk base
  const int boff = (lane < 16) ? 0 : 8;    // B fragment k-chunk base
  const int rbase = (lane < 16) ? 0 : 8;   // C fragment row base

  // async W panel copy: 64 rows x 64 k x 2B = 8KB = 512 x 16B, 2 chunks/thread
  const int c0   = threadIdx.x;            // chunk ids c0, c0+256
  const int wr0  = c0 >> 3, wc0 = (c0 & 7) * 8;
  const int wr1  = (c0 + 256) >> 3, wc1 = ((c0 + 256) & 7) * 8;

  auto loadW = [&](int buf, int k0) {
    CMA_ASYNC_COPY16(&sW[buf][wr0 * 64 + wc0],
                     W + (size_t)(nb + wr0) * K + k0 + wc0);
    CMA_ASYNC_COPY16(&sW[buf][wr1 * 64 + wc1],
                     W + (size_t)(nb + wr1) * K + k0 + wc1);
  };

  const unsigned short* ap = A + (size_t)(mt + l16) * K;
  auto loadA = [&](Frag32* f, int k0) {
#pragma unroll
    for (int kk = 0; kk < 2; ++kk) {
      f[kk].q[0] = *(const i4*)(ap + k0 + kk * 32 + aoff);
      f[kk].q[1] = *(const i4*)(ap + k0 + kk * 32 + aoff + 16);
    }
  };

  v8f acc[4] = {};
  Frag32 aC[2], aN[2];

  auto compute = [&](int buf, Frag32* af) {
#pragma unroll
    for (int kk = 0; kk < 2; ++kk) {
#pragma unroll
      for (int nt = 0; nt < 4; ++nt) {
        const unsigned short* wp = &sW[buf][(nt * 16 + l16) * 64 + kk * 32 + boff];
        Frag32 b;
        b.q[0] = *(const i4*)(wp);
        b.q[1] = *(const i4*)(wp + 16);
        acc[nt] = wmma_bf16(af[kk].v, b.v, acc[nt]);
      }
    }
  };

  loadW(0, 0);
  loadA(aC, 0);
  __builtin_prefetch(ap + 128, 0, 0);

  int buf = 0;
  int k0 = 0;
  for (; k0 + 64 < K; k0 += 64, buf ^= 1) {   // branchless steady state
    loadW(buf ^ 1, k0 + 64);                  // prefetch next stage
    loadA(aN, k0 + 64);
    __builtin_prefetch(ap + k0 + 192, 0, 0);
    CMA_ASYNC_WAIT(2);                        // drain current stage only
    __syncthreads();
    compute(buf, aC);
    aC[0] = aN[0];                            // unconditional register rotate
    aC[1] = aN[1];
    __syncthreads();                          // reads done before overwrite
  }
  CMA_ASYNC_WAIT(0);                          // final stage
  __syncthreads();
  compute(buf, aC);

  if (mode == 0) {
    unsigned short* ob = (unsigned short*)out;
#pragma unroll
    for (int nt = 0; nt < 4; ++nt) {
      int n = nb + nt * 16 + l16;
      float bn = bias[n];
#pragma unroll
      for (int r = 0; r < 8; ++r) {
        int m = mt + rbase + r;
        ob[(size_t)m * N + n] = f32_to_bf16(acc[nt][r] + bn);
      }
    }
  } else if (mode == 1) {
    // Vt[b][h][d][p] : rows (b,p) of C become contiguous p-runs -> 128-bit stores
    unsigned short* ob = (unsigned short*)out;
    const int p0 = (mt & 1023) + rbase;
    const int bb = mt >> 10;
#pragma unroll
    for (int nt = 0; nt < 4; ++nt) {
      int n = nb + nt * 16 + l16;
      float bn = bias[n];
      int h = n / 96, d = n % 96;
      Pack8 pk;
#pragma unroll
      for (int r = 0; r < 8; ++r) pk.s[r] = f32_to_bf16(acc[nt][r] + bn);
      *(i4*)(ob + (((size_t)(bb * 8 + h) * 96 + d) << 10) + p0) = pk.q;
    }
  } else {
    float* of = (float*)out;
#pragma unroll
    for (int nt = 0; nt < 4; ++nt) {
      int n = nb + nt * 16 + l16;
      float bn = bias[n];
#pragma unroll
      for (int r = 0; r < 8; ++r) {
        int m = mt + rbase + r;
        of[(size_t)m * N + n] = acc[nt][r] + bn;
      }
    }
  }
}

// ---------------------------------------------------------------------------
// Attention core. One workgroup per (b, 16-row m-tile); loops over 8 heads.
//   Phase 1: S[16x1024] = (Q K^T) * rsqrt(96) into LDS via WMMA
//   Phase 2: exact softmax per row; stream P/H into attn_avg (owned slab)
//   Phase 3: pack P as bf16 in LDS, O += P V via WMMA against Vt[b][h][d][p]
//   Phase 4: cross-wave K-partial reduction of O through LDS -> bf16 attended
// ---------------------------------------------------------------------------
__global__ __launch_bounds__(256)
void cma_attn(const unsigned short* __restrict__ Qb,   // [8,1024,768] bf16
              const unsigned short* __restrict__ Kb,   // [8,1024,768] bf16
              const unsigned short* __restrict__ Vt,   // [8,8,96,1024] bf16
              unsigned short* __restrict__ Ao,         // [8,1024,768] bf16
              float* __restrict__ avg_out) {           // [8,1024,1024] f32
  __shared__ float sS[16 * 1024];                      // 64 KB score panel

  const int b    = blockIdx.y;
  const int m0   = blockIdx.x * 16;
  const int lane = threadIdx.x & 31;
  const int wave = threadIdx.x >> 5;
  const int l16  = lane & 15;
  const int aoff = (lane < 16) ? 0 : 8;
  const int boff = (lane < 16) ? 0 : 8;
  const int rbase = (lane < 16) ? 0 : 8;
  const float scale = 0.1020620726f;                   // 1/sqrt(96)

  unsigned short* sP = (unsigned short*)sS;            // bf16 overlay (low 32KB)

  for (int h = 0; h < 8; ++h) {
    // ----- Phase 1: scores -> LDS --------------------------------------
    Frag32 aQ[3];
    {
      const unsigned short* qp =
          Qb + ((size_t)((b << 10) + m0 + l16) * 768) + h * 96;
#pragma unroll
      for (int kk = 0; kk < 3; ++kk) {
        aQ[kk].q[0] = *(const i4*)(qp + kk * 32 + aoff);
        aQ[kk].q[1] = *(const i4*)(qp + kk * 32 + aoff + 16);
      }
    }
#pragma unroll
    for (int t = 0; t < 8; ++t) {
      const int p0 = wave * 128 + t * 16;
      const unsigned short* kp =
          Kb + ((size_t)((b << 10) + p0 + l16) * 768) + h * 96 + boff;
      v8f s = {};
#pragma unroll
      for (int kk = 0; kk < 3; ++kk) {
        Frag32 bk;
        bk.q[0] = *(const i4*)(kp + kk * 32);
        bk.q[1] = *(const i4*)(kp + kk * 32 + 16);
        s = wmma_bf16(aQ[kk].v, bk.v, s);
      }
      const int ncol = p0 + l16;
#pragma unroll
      for (int r = 0; r < 8; ++r)
        sS[(rbase + r) * 1024 + ncol] = s[r] * scale;
    }
    __syncthreads();

    // ----- Phase 2: softmax (2 rows / wave), attn_avg accumulation -----
    float P[2][32];
#pragma unroll
    for (int rr = 0; rr < 2; ++rr) {
      const int row = wave * 2 + rr;
      float mx = -3.0e38f;
#pragma unroll
      for (int i = 0; i < 32; ++i) {
        P[rr][i] = sS[row * 1024 + lane + 32 * i];
        mx = fmaxf(mx, P[rr][i]);
      }
#pragma unroll
      for (int off = 16; off >= 1; off >>= 1)
        mx = fmaxf(mx, __shfl_xor(mx, off));
      float sum = 0.f;
#pragma unroll
      for (int i = 0; i < 32; ++i) {
        float e = __expf(P[rr][i] - mx);
        P[rr][i] = e;
        sum += e;
      }
#pragma unroll
      for (int off = 16; off >= 1; off >>= 1)
        sum += __shfl_xor(sum, off);
      const float inv = 1.0f / sum;
      float* ap = avg_out + ((size_t)((b << 10) + m0 + row) << 10) + lane;
#pragma unroll
      for (int i = 0; i < 32; ++i) {
        P[rr][i] *= inv;
        float v = P[rr][i] * 0.125f;                   // mean over 8 heads
        if (h == 0) ap[32 * i] = v;                    // init over poison
        else        ap[32 * i] += v;                   // WG-exclusive slab
      }
    }
    __syncthreads();                                   // all S reads done
#pragma unroll
    for (int rr = 0; rr < 2; ++rr) {
      const int row = wave * 2 + rr;
#pragma unroll
      for (int i = 0; i < 32; ++i)
        sP[row * 1024 + lane + 32 * i] = f32_to_bf16(P[rr][i]);
    }
    __syncthreads();

    // ----- Phase 3: O_partial += P * V over this wave's 128-wide K slab -
    v8f o[6] = {};
#pragma unroll
    for (int c4 = 0; c4 < 4; ++c4) {
      const int pb = wave * 128 + c4 * 32;
      Frag32 aP;
      const unsigned short* pp = sP + (size_t)l16 * 1024 + pb;
      aP.q[0] = *(const i4*)(pp + aoff);
      aP.q[1] = *(const i4*)(pp + aoff + 16);
#pragma unroll
      for (int nt = 0; nt < 6; ++nt) {
        const unsigned short* vp =
            Vt + (((size_t)(b * 8 + h) * 96 + nt * 16 + l16) << 10) + pb + boff;
        Frag32 bv;
        bv.q[0] = *(const i4*)(vp);
        bv.q[1] = *(const i4*)(vp + 16);
        o[nt] = wmma_bf16(aP.v, bv.v, o[nt]);
      }
    }
    __syncthreads();                                   // P reads done; reuse LDS

    // ----- Phase 4: reduce 8 wave-partials of O[16x96] through LDS ------
#pragma unroll
    for (int nt = 0; nt < 6; ++nt)
#pragma unroll
      for (int r = 0; r < 8; ++r)
        sS[wave * 1536 + (rbase + r) * 96 + nt * 16 + l16] = o[nt][r];
    __syncthreads();
    for (int idx = threadIdx.x; idx < 1536; idx += 256) {
      float sum = 0.f;
#pragma unroll
      for (int w = 0; w < 8; ++w) sum += sS[w * 1536 + idx];
      const int m = idx / 96, d = idx % 96;
      Ao[((size_t)((b << 10) + m0 + m) * 768) + h * 96 + d] = f32_to_bf16(sum);
    }
    __syncthreads();
  }
}

// ---------------------------------------------------------------------------
extern "C" void kernel_launch(void* const* d_in, const int* in_sizes, int n_in,
                              void* d_out, int out_size, void* d_ws, size_t ws_size,
                              hipStream_t stream) {
  const float* mam = (const float*)d_in[0];
  const float* pat = (const float*)d_in[1];
  const float* qw  = (const float*)d_in[2];
  const float* qb  = (const float*)d_in[3];
  const float* kw  = (const float*)d_in[4];
  const float* kb  = (const float*)d_in[5];
  const float* vw  = (const float*)d_in[6];
  const float* vb  = (const float*)d_in[7];
  const float* ow  = (const float*)d_in[8];
  const float* ob  = (const float*)d_in[9];

  const size_t TOK = 8192ull * 768ull;   // 6,291,456 elements
  const size_t WSZ = 768ull * 768ull;    //   589,824 elements

  unsigned short* ws = (unsigned short*)d_ws;
  unsigned short* Xm = ws;               // mammo bf16  [8192,768]
  unsigned short* Xp = Xm + TOK;         // patho bf16  [8192,768]
  unsigned short* Wq = Xp + TOK;
  unsigned short* Wk = Wq + WSZ;
  unsigned short* Wv = Wk + WSZ;
  unsigned short* Wo = Wv + WSZ;
  unsigned short* Qb = Wo + WSZ;         // Q bf16      [8192,768]
  unsigned short* Kb = Qb + TOK;         // K bf16      [8192,768]
  unsigned short* Vt = Kb + TOK;         // V bf16 transposed [8,8,96,1024]
  unsigned short* Ao = Vt + TOK;         // attended pre-proj bf16 [8192,768]

  float* att_out = (float*)d_out;        // [8,1024,768]
  float* avg_out = att_out + TOK;        // [8,1024,1024]

  cma_cvt_bf16<<<2048, 256, 0, stream>>>(mam, Xm, (int)TOK);
  cma_cvt_bf16<<<2048, 256, 0, stream>>>(pat, Xp, (int)TOK);
  cma_cvt_bf16<<<512, 256, 0, stream>>>(qw, Wq, (int)WSZ);
  cma_cvt_bf16<<<512, 256, 0, stream>>>(kw, Wk, (int)WSZ);
  cma_cvt_bf16<<<512, 256, 0, stream>>>(vw, Wv, (int)WSZ);
  cma_cvt_bf16<<<512, 256, 0, stream>>>(ow, Wo, (int)WSZ);

  dim3 gg(64, 12);
  cma_gemm_tn<<<gg, 256, 0, stream>>>(Xm, Wq, qb, Qb, 8192, 768, 768, 0);
  cma_gemm_tn<<<gg, 256, 0, stream>>>(Xp, Wk, kb, Kb, 8192, 768, 768, 0);
  cma_gemm_tn<<<gg, 256, 0, stream>>>(Xp, Wv, vb, Vt, 8192, 768, 768, 1);

  cma_attn<<<dim3(64, 8), 256, 0, stream>>>(Qb, Kb, Vt, Ao, avg_out);

  cma_gemm_tn<<<gg, 256, 0, stream>>>(Ao, Wo, ob, att_out, 8192, 768, 768, 2);
}
